// LSTM_model_10496900071757
// MI455X (gfx1250) — matrix operand order repaired and checked by reading, not compiled, not run on hardware
//
#include <hip/hip_runtime.h>
#include <hip/hip_bf16.h>
#include <math.h>

#define VOCAB 32000
#define EMBED 512
#define NH    1024
#define NG    4096   /* 4*NH */
#define BATCH 64
#define SEQ   512
#define NOUT  8
#define NWG   64     /* persistent workgroups for the LSTM scan */

typedef __attribute__((ext_vector_type(16))) __bf16 v16bf;
typedef __attribute__((ext_vector_type(8)))  float  v8f;

union Frag {
  v16bf  v;
  uint4  q[2];
  __bf16 h[16];
};

__device__ __forceinline__ float sigmoidf_(float x) {
  return 1.0f / (1.0f + __expf(-x));
}

// ---- gfx1250 async global->LDS copy (ASYNCcnt) ----------------------------
#if defined(__has_builtin)
#if __has_builtin(__builtin_amdgcn_global_load_async_to_lds_b128) && \
    __has_builtin(__builtin_amdgcn_s_wait_asynccnt)
#define USE_ASYNC_LDS 1
#endif
#endif

#ifdef USE_ASYNC_LDS
typedef int v4i_t __attribute__((ext_vector_type(4)));
typedef __attribute__((address_space(1))) v4i_t* as1_v4i;
typedef __attribute__((address_space(3))) v4i_t* as3_v4i;
__device__ __forceinline__ void async_copy16(const void* g, void* l) {
  // Generic->AS1 is a value-preserving 64b address; generic LDS address low
  // 32 bits are the hardware LDS offset (ISA 10.2 aperture rules).
  __builtin_amdgcn_global_load_async_to_lds_b128(
      (as1_v4i)(unsigned long long)g,
      (as3_v4i)(unsigned)(unsigned long long)l, 0, 0);
}
#endif

// ---------------------------------------------------------------------------
// Kernel 1: pre-swizzle a fp32 [KROWS, NG] weight matrix into bf16 WMMA
// B-fragment layout. B-frag, contiguous-k order used here (A matches): lane = (n&15) + 16*(k'>>4),
// byte pos = (k'&15)*2, so 16 consecutive K for a fixed n map to one
// contiguous 32B lane chunk. Chunk index = (ntG*(KROWS/32) + ktile)*32 + lane.
// ---------------------------------------------------------------------------
template <int KROWS>
__global__ __launch_bounds__(256) void k_swz(const float* __restrict__ src,
                                             __bf16* __restrict__ dst) {
  constexpr int KT = KROWS / 32;
  int idx = blockIdx.x * 256 + threadIdx.x;   // NG * (KROWS/16) threads
  int n   = idx & (NG - 1);
  int kk  = idx >> 12;                        // 16-K half-chunk index
  int ntG   = n >> 4;
  int ktile = kk >> 1;
  int khalf = kk & 1;
  int lane  = (n & 15) + 16 * khalf;
  Frag f;
  const float* s = src + (size_t)(kk * 16) * NG + n;
#pragma unroll
  for (int i = 0; i < 16; ++i) f.h[i] = (__bf16)s[(size_t)i * NG];
  uint4* d = (uint4*)(dst + (((size_t)ntG * KT + ktile) * 32 + lane) * 16);
  d[0] = f.q[0];
  d[1] = f.q[1];
}

// ---------------------------------------------------------------------------
// Kernel 2: xW = emb[x] @ W + hid_bias   (M=B*S=32768, K=512, N=4096), bf16
// WMMA, fp32 accumulate. 64x64 tile/WG, 8 waves; wave owns mtile = w&3 and
// ntiles {w>>2, (w>>2)+2}. A (embedding gather) is converted + swizzled
// through LDS (one b128 store per thread per iter); B fragments come straight
// from the pre-swizzled bf16 W (L2-resident) via global_load_b128.
// ---------------------------------------------------------------------------
__global__ __launch_bounds__(256) void k_xw(const int*    __restrict__ x,
                                            const float*  __restrict__ emb,
                                            const __bf16* __restrict__ wswz,
                                            const float*  __restrict__ bias,
                                            float* __restrict__ xW) {
  __shared__ uint4 aFrag[4][32][2];                         // 4 KB
  const int tid  = threadIdx.x;
  const int lane = tid & 31;
  const int w    = tid >> 5;
  const int mt   = w & 3;
  const int half = w >> 2;
  const int nt0  = half, nt1 = half + 2;
  const int mBase = blockIdx.x * 64;
  const int nBase = blockIdx.y * 64;
  const int ntG0  = (nBase >> 4) + nt0;       // global 16-col tile indices
  const int ntG1  = (nBase >> 4) + nt1;

  // A staging role: row rA (0..63), 8-K chunk kc (0..3)
  const int rA    = tid >> 2;
  const int kc    = tid & 3;
  const int aLane = (rA & 15) + 16 * (kc >> 1);
  const int aSlot = kc & 1;
  const int aMt   = rA >> 4;
  const int token = x[mBase + rA];
  const float* aSrc = emb + (size_t)token * EMBED + kc * 8;

  v8f acc0 = {}; v8f acc1 = {};

  for (int kt = 0; kt < EMBED / 32; ++kt) {
    const int k0 = kt * 32;
    {  // stage A fragment bytes (gathered embedding rows, fp32 -> bf16)
      const float4* s = (const float4*)(aSrc + k0);
      float4 f0 = s[0], f1 = s[1];
      Frag fa;
      fa.h[0]=(__bf16)f0.x; fa.h[1]=(__bf16)f0.y; fa.h[2]=(__bf16)f0.z; fa.h[3]=(__bf16)f0.w;
      fa.h[4]=(__bf16)f1.x; fa.h[5]=(__bf16)f1.y; fa.h[6]=(__bf16)f1.z; fa.h[7]=(__bf16)f1.w;
      aFrag[aMt][aLane][aSlot] = fa.q[0];
    }
    __syncthreads();
    Frag fa, fb0, fb1;
    fa.q[0] = aFrag[mt][lane][0];
    fa.q[1] = aFrag[mt][lane][1];
    const uint4* b0 = (const uint4*)(wswz + (((size_t)ntG0 * (EMBED / 32) + kt) * 32 + lane) * 16);
    const uint4* b1 = (const uint4*)(wswz + (((size_t)ntG1 * (EMBED / 32) + kt) * 32 + lane) * 16);
    fb0.q[0] = b0[0]; fb0.q[1] = b0[1];
    fb1.q[0] = b1[0]; fb1.q[1] = b1[1];
    acc0 = __builtin_amdgcn_wmma_f32_16x16x32_bf16(false, fa.v, false, fb0.v,
                                                   (short)0, acc0, false, false);
    acc1 = __builtin_amdgcn_wmma_f32_16x16x32_bf16(false, fa.v, false, fb1.v,
                                                   (short)0, acc1, false, false);
    __syncthreads();
  }

  const int hi = lane >> 4;
  const int cl = lane & 15;
#pragma unroll
  for (int v = 0; v < 8; ++v) {
    int row = mBase + mt * 16 + v + 8 * hi;
    int c0  = nBase + nt0 * 16 + cl;
    int c1  = nBase + nt1 * 16 + cl;
    xW[(size_t)row * NG + c0] = acc0[v] + bias[c0];
    xW[(size_t)row * NG + c1] = acc1[v] + bias[c1];
  }
}

// ---------------------------------------------------------------------------
// Kernel 3: persistent LSTM scan. 64 WGs; WG `wg` owns h columns
// [wg*16, wg*16+16) + the 4 matching gate slices. Per step each wave runs a
// 16x1024 @ 1024x32 bf16 WMMA strip; the step's 16KB xW slice is copied to
// LDS with global_load_async_to_lds_b128 overlapping the WMMA loop
// (s_wait_asynccnt before use). Gates go through LDS for the fp32
// nonlinearity + cell update; new h (bf16) is double-buffered in global;
// a monotonic-counter grid barrier with threadfence release/acquire
// (global_wb / global_inv) gates each step.
// ---------------------------------------------------------------------------
__global__ __launch_bounds__(256) void k_lstm(const float*  __restrict__ xW,
                                              const __bf16* __restrict__ uswz,
                                              __bf16* __restrict__ h0buf,
                                              __bf16* __restrict__ h1buf,
                                              float*  __restrict__ hOut,
                                              unsigned* __restrict__ cnt) {
  __shared__ float gatesLDS[64][68];   // +pad to dodge bank conflicts
  __shared__ float cLDS[64][16];
#ifdef USE_ASYNC_LDS
  __shared__ float xwLDS[64][64];      // this step's xW gate slice (16 KB)
#endif
  const int tid  = threadIdx.x;
  const int lane = tid & 31;
  const int w    = tid >> 5;
  const int mt   = w & 3;
  const int half = w >> 2;
  const int q0 = half, q1 = half + 2;
  const int wg = blockIdx.x;

  for (int i = tid; i < 64 * 16; i += 256) cLDS[i >> 4][i & 15] = 0.0f;
  __syncthreads();

  const int hi   = lane >> 4;
  const int cl   = lane & 15;
  const int rowA = mt * 16 + cl;          // A-matrix row this lane loads
  const int ntG0 = q0 * 64 + wg;          // global B-fragment column tiles
  const int ntG1 = q1 * 64 + wg;

  unsigned target = 0;
  for (int t = 0; t < SEQ; ++t) {
    const __bf16* hIn   = (t & 1) ? h1buf : h0buf;
    __bf16*       hNext = (t & 1) ? h0buf : h1buf;

#ifdef USE_ASYNC_LDS
    // Kick off the async xW slice copy; it completes under the WMMA loop.
#pragma unroll
    for (int k = 0; k < 4; ++k) {
      int tau = tid + k * 256;            // 1024 x 16B lane-tasks = 16 KB
      int b   = tau >> 4;
      int rem = tau & 15;
      int q   = rem >> 2;
      int sub = rem & 3;
      const float* g = xW + ((size_t)b * SEQ + t) * NG + q * NH + wg * 16 + sub * 4;
      async_copy16(g, &xwLDS[b][q * 16 + sub * 4]);
    }
#else
    {  // prefetch this step's xW gate operands (global_prefetch_b8)
      int id = tid * 4;
      const float* p = xW + ((size_t)(id >> 4) * SEQ + t) * NG + wg * 16 + (id & 15);
      __builtin_prefetch(p, 0, 1);
      __builtin_prefetch(p + NH, 0, 1);
      __builtin_prefetch(p + 2 * NH, 0, 1);
      __builtin_prefetch(p + 3 * NH, 0, 1);
    }
#endif

    v8f acc0 = {}; v8f acc1 = {};
    for (int kt = 0; kt < NH / 32; ++kt) {
      // A fragment straight from row-major bf16 h (two b128 per lane)
      const uint4* hp = (const uint4*)(hIn + (size_t)rowA * NH + kt * 32 + hi * 16);
      Frag fa; fa.q[0] = hp[0]; fa.q[1] = hp[1];
      // B fragments from pre-swizzled U (L2-resident)
      const uint4* u0 = (const uint4*)(uswz + (((size_t)ntG0 * 32 + kt) * 32 + lane) * 16);
      const uint4* u1 = (const uint4*)(uswz + (((size_t)ntG1 * 32 + kt) * 32 + lane) * 16);
      Frag fb0; fb0.q[0] = u0[0]; fb0.q[1] = u0[1];
      Frag fb1; fb1.q[0] = u1[0]; fb1.q[1] = u1[1];
      acc0 = __builtin_amdgcn_wmma_f32_16x16x32_bf16(false, fa.v, false, fb0.v,
                                                     (short)0, acc0, false, false);
      acc1 = __builtin_amdgcn_wmma_f32_16x16x32_bf16(false, fa.v, false, fb1.v,
                                                     (short)0, acc1, false, false);
    }

#pragma unroll
    for (int v = 0; v < 8; ++v) {
      int row = mt * 16 + v + 8 * hi;
      gatesLDS[row][q0 * 16 + cl] = acc0[v];
      gatesLDS[row][q1 * 16 + cl] = acc1[v];
    }
#ifdef USE_ASYNC_LDS
    __builtin_amdgcn_s_wait_asynccnt(0);   // xW slice now resident in LDS
#endif
    __syncthreads();

#pragma unroll
    for (int i = 0; i < 4; ++i) {
      int id  = tid * 4 + i;
      int b   = id >> 4;
      int col = id & 15;
#ifdef USE_ASYNC_LDS
      float gi = gatesLDS[b][col]      + xwLDS[b][col];
      float gf = gatesLDS[b][16 + col] + xwLDS[b][16 + col];
      float gg = gatesLDS[b][32 + col] + xwLDS[b][32 + col];
      float go = gatesLDS[b][48 + col] + xwLDS[b][48 + col];
#else
      const float* xwp = xW + ((size_t)b * SEQ + t) * NG + wg * 16 + col;
      float gi = gatesLDS[b][col]      + xwp[0];
      float gf = gatesLDS[b][16 + col] + xwp[NH];
      float gg = gatesLDS[b][32 + col] + xwp[2 * NH];
      float go = gatesLDS[b][48 + col] + xwp[3 * NH];
#endif
      float ig = sigmoidf_(gi);
      float fg = sigmoidf_(gf);
      float gv = tanhf(gg);
      float og = sigmoidf_(go);
      float c  = fg * cLDS[b][col] + ig * gv;
      cLDS[b][col] = c;
      float h = og * tanhf(c);
      hNext[(size_t)b * NH + wg * 16 + col] = (__bf16)h;
      if (t == SEQ - 1) hOut[(size_t)b * NH + wg * 16 + col] = h;
    }

    // ---- device-wide barrier (release: global_wb; acquire: global_inv) ----
    __threadfence();
    __syncthreads();
    target += NWG;
    if (tid == 0) {
      atomicAdd(cnt, 1u);
      while (__hip_atomic_load(cnt, __ATOMIC_RELAXED, __HIP_MEMORY_SCOPE_AGENT) < target) {
        __builtin_amdgcn_s_sleep(1);
      }
    }
    __syncthreads();
    __threadfence();
  }
}

// ---------------------------------------------------------------------------
// Kernel 4: out = h_final @ V + out_bias  (64x1024 @ 1024x8, tiny, fp32)
// ---------------------------------------------------------------------------
__global__ __launch_bounds__(256) void k_out(const float* __restrict__ h,
                                             const float* __restrict__ V,
                                             const float* __restrict__ ob,
                                             float* __restrict__ out) {
  int idx = blockIdx.x * 256 + threadIdx.x;
  if (idx >= BATCH * NOUT) return;
  int b = idx >> 3, o = idx & 7;
  float s = 0.f;
  for (int k = 0; k < NH; ++k) s += h[(size_t)b * NH + k] * V[(size_t)k * NOUT + o];
  out[idx] = s + ob[o];
}

extern "C" void kernel_launch(void* const* d_in, const int* in_sizes, int n_in,
                              void* d_out, int out_size, void* d_ws, size_t ws_size,
                              hipStream_t stream) {
  (void)in_sizes; (void)n_in; (void)out_size; (void)ws_size;
  const int*   x        = (const int*)d_in[0];
  const float* emb      = (const float*)d_in[1];
  const float* W        = (const float*)d_in[2];
  const float* U        = (const float*)d_in[3];
  const float* hid_bias = (const float*)d_in[4];
  const float* V        = (const float*)d_in[5];
  const float* out_bias = (const float*)d_in[6];
  float* out = (float*)d_out;

  char* ws = (char*)d_ws;
  size_t off = 0;
  float*  xW   = (float*)(ws + off);  off += (size_t)BATCH * SEQ * NG * sizeof(float); // 512 MB
  __bf16* uswz = (__bf16*)(ws + off); off += (size_t)NH * NG * sizeof(__bf16);         // 8 MB
  __bf16* wswz = (__bf16*)(ws + off); off += (size_t)EMBED * NG * sizeof(__bf16);      // 4 MB
  char*   zbase = ws + off;
  __bf16* h0   = (__bf16*)(ws + off); off += (size_t)BATCH * NH * sizeof(__bf16);      // 128 KB
  __bf16* h1   = (__bf16*)(ws + off); off += (size_t)BATCH * NH * sizeof(__bf16);
  float*  hF   = (float*)(ws + off);  off += (size_t)BATCH * NH * sizeof(float);
  unsigned* cnt = (unsigned*)(ws + off); off += 256;
  size_t zsize = (size_t)((ws + off) - zbase);

  (void)hipMemsetAsync(zbase, 0, zsize, stream);         // zero h0/h1/hF/counter

  k_swz<NH>   <<<(NG * (NH / 16))    / 256, 256, 0, stream>>>(U, uswz);   // 1024 blocks
  k_swz<EMBED><<<(NG * (EMBED / 16)) / 256, 256, 0, stream>>>(W, wswz);   // 512 blocks

  dim3 g2(BATCH * SEQ / 64, NG / 64);                    // (512, 64)
  k_xw<<<g2, 256, 0, stream>>>(x, emb, wswz, hid_bias, xW);

  k_lstm<<<NWG, 256, 0, stream>>>(xW, uswz, h0, h1, hF, cnt);

  k_out<<<2, 256, 0, stream>>>(hF, V, out_bias, out);
}
